// MambaBlock_22514218565766
// MI455X (gfx1250) — matrix operand order repaired and checked by reading, not compiled, not run on hardware
//
#include <hip/hip_runtime.h>
#include <cstddef>
#include <cstdint>

// ---------------------------------------------------------------------------
// Shapes (fixed for this problem)
// ---------------------------------------------------------------------------
#define B_N      4
#define L_N      2048
#define D_M      768
#define D_INNER  1536
#define D_STATE  128
#define CONV_DIM 1792
#define D_IN_PROJ 3329
#define D_FF     3072
#define ROWS     (B_N * L_N)   // 8192

// Toggle: async global->LDS staging of the A tile (gfx1250 ASYNCcnt path).
#define USE_ASYNC_LDS 1

// ---------------------------------------------------------------------------
// WMMA types (CDNA5 / gfx1250, wave32)
// ---------------------------------------------------------------------------
typedef __attribute__((ext_vector_type(16))) __bf16          v16bf;
typedef __attribute__((ext_vector_type(8)))  float           v8f;
typedef __attribute__((ext_vector_type(8)))  unsigned short  us8;

union Frag { v16bf v; us8 h[2]; };

// fp32 -> bf16 round-to-nearest-even
__device__ __forceinline__ unsigned short f2bf(float f) {
  unsigned int u = __float_as_uint(f);
  unsigned int r = u + 0x7FFFu + ((u >> 16) & 1u);
  return (unsigned short)(r >> 16);
}

// ---------------------------------------------------------------------------
// fp32 -> bf16 buffer convert (weights / activations)
// ---------------------------------------------------------------------------
__global__ void f32_to_bf16_kernel(const float* __restrict__ in,
                                   unsigned short* __restrict__ out, size_t n) {
  size_t i = (size_t)blockIdx.x * 256 + threadIdx.x;
  if (i < n) out[i] = f2bf(in[i]);
}

// ---------------------------------------------------------------------------
// RMSNorm (optionally fused y*silu(z) gate) -> bf16 output
// one workgroup (256 threads) per row
// ---------------------------------------------------------------------------
__global__ void rmsnorm_kernel(const float* __restrict__ x, int xld,
                               const float* __restrict__ zg, int zld,
                               const float* __restrict__ w,
                               unsigned short* __restrict__ out,
                               int D, int gate) {
  const int row = blockIdx.x;
  const int tid = threadIdx.x;
  __shared__ float red[256];
  const float* xr = x + (size_t)row * xld;
  const float* zr = zg + (size_t)row * zld;

  float s = 0.f;
  for (int i = tid; i < D; i += 256) {
    float v = xr[i];
    if (gate) { float z = zr[i]; v *= z / (1.f + __expf(-z)); }
    s += v * v;
  }
  red[tid] = s;
  __syncthreads();
  for (int off = 128; off > 0; off >>= 1) {
    if (tid < off) red[tid] += red[tid + off];
    __syncthreads();
  }
  const float scale = rsqrtf(red[0] / (float)D + 1e-5f);
  for (int i = tid; i < D; i += 256) {
    float v = xr[i];
    if (gate) { float z = zr[i]; v *= z / (1.f + __expf(-z)); }
    out[(size_t)row * D + i] = f2bf(v * scale * w[i]);
  }
}

// ---------------------------------------------------------------------------
// bf16 WMMA GEMM:  C[M,N] = A[M,K] @ B[K,N]   (fp32 accumulate)
//   MODE 0: Cf = A@B
//   MODE 1: Cf = A@B + Rf                (residual add)
//   MODE 2: Cb = bf16(gelu_exact(A@B))   (MLP1 epilogue)
// Workgroup: 256 threads (8 waves) -> 128x128 tile; wave -> 32x64 sub-tile
// via 2x4 grid of v_wmma_f32_16x16x32_bf16. K-step = 32 through LDS.
// A tile staged global->LDS with global_load_async_to_lds_b128 (ASYNCcnt).
// Requires: M % 128 == 0, K % 32 == 0 (true for all calls); N edge guarded.
// ---------------------------------------------------------------------------
template <int MODE>
__global__ __launch_bounds__(256, 2)
void gemm_wmma_bf16(const unsigned short* __restrict__ Ag,
                    const unsigned short* __restrict__ Bg,
                    float* __restrict__ Cf,
                    unsigned short* __restrict__ Cb,
                    const float* __restrict__ Rf,
                    int M, int N, int K) {
  // +8-ushort pad keeps every fragment sub-row 16B aligned and de-banks LDS
  __shared__ unsigned short lA[128][40];   // A tile, row-major [m][k]
  __shared__ unsigned short lB[128][40];   // B tile, TRANSPOSED [n][k]

  const int tid  = threadIdx.x;
  const int lane = tid & 31;
  const int wave = tid >> 5;
  const int wm   = wave >> 1;          // 0..3 : 32-row strip
  const int wn   = wave & 1;           // 0..1 : 64-col strip
  const int bm   = blockIdx.y * 128;
  const int bn   = blockIdx.x * 128;
  const bool fullN = (bn + 128 <= N); // wave-uniform: folds the edge guards

  v8f acc[2][4] = {};

  const int arow = tid >> 1;           // 0..127
  const int acol = (tid & 1) * 16;     // 0 or 16

  for (int kb = 0; kb < K; kb += 32) {
    // ---- stage A tile (two 16B chunks per thread; K%32==0 => aligned) ----
    {
      const unsigned short* gp = Ag + (size_t)(bm + arow) * K + kb + acol;
#if USE_ASYNC_LDS
      // Direct global->LDS async copy; low 32 bits of a generic shared
      // pointer are the LDS byte offset on gfx1250.
      unsigned la = (unsigned)(size_t)(void*)&lA[arow][acol];
      asm volatile("global_load_async_to_lds_b128 %0, %1, off"
                   :: "v"(la), "v"(gp) : "memory");
      asm volatile("global_load_async_to_lds_b128 %0, %1, off"
                   :: "v"(la + 16u), "v"(gp + 8) : "memory");
#else
      const us8* g = (const us8*)gp;
      *(us8*)&lA[arow][acol]     = g[0];
      *(us8*)&lA[arow][acol + 8] = g[1];
#endif
    }
    // ---- stage B tile transposed into lB[n][k] ----
    if (fullN) {
      for (int r = 0; r < 16; ++r) {
        int f = r * 256 + tid;           // 0..4095
        int k = f >> 7;                  // 0..31
        int n = f & 127;                 // 0..127
        lB[n][k] = Bg[(size_t)(kb + k) * N + bn + n];
      }
    } else {
      for (int r = 0; r < 16; ++r) {
        int f = r * 256 + tid;
        int k = f >> 7;
        int n = f & 127;
        unsigned short v = 0;
        int gn = bn + n;
        if (gn < N) v = Bg[(size_t)(kb + k) * N + gn];
        lB[n][k] = v;
      }
    }
    if (kb + 32 < K)   // gfx1250 global_prefetch_b8 on next B tile
      __builtin_prefetch(Bg + (size_t)(kb + 32) * N + bn + tid, 0, 1);
#if USE_ASYNC_LDS
    asm volatile("s_wait_asynccnt 0x0" ::: "memory");
#endif
    __syncthreads();

    // ---- fragments ----
    // A 16x32 bf16 layout: lane m = lane&15; lanes>=16 take the K+8 halves.
    Frag fa[2], fb[4];
    const int m0 = wm * 32 + (lane & 15);
    const int kh = (lane >> 4) * 8;      // 0 or 8
    #pragma unroll
    for (int mi = 0; mi < 2; ++mi) {
      fa[mi].h[0] = *(const us8*)&lA[m0 + mi * 16][kh];        // K kh..kh+7
      fa[mi].h[1] = *(const us8*)&lA[m0 + mi * 16][16 + kh];   // K 16+kh..
    }
    // B 32x16 bf16 layout: lane n = lane&15; lanes>=16 take the K+8 halves.
    const int n0  = wn * 64 + (lane & 15);
    const int kbq = (lane >> 4) * 8;     // 0 or 8
    #pragma unroll
    for (int ni = 0; ni < 4; ++ni) {
      const us8* q = (const us8*)&lB[n0 + ni * 16][kbq];
      fb[ni].h[0] = q[0];
      fb[ni].h[1] = q[2];
    }

    #pragma unroll
    for (int mi = 0; mi < 2; ++mi)
      #pragma unroll
      for (int ni = 0; ni < 4; ++ni)
        acc[mi][ni] = __builtin_amdgcn_wmma_f32_16x16x32_bf16(
            false, fa[mi].v, false, fb[ni].v, (short)0, acc[mi][ni],
            false, false);
    __syncthreads();
  }

  // ---- epilogue: D layout VGPR j -> M=(lane>>4)*8+j, N=lane&15 ----
  auto store_tile = [&](bool guard) {
    #pragma unroll
    for (int mi = 0; mi < 2; ++mi)
      #pragma unroll
      for (int ni = 0; ni < 4; ++ni)
        #pragma unroll
        for (int j = 0; j < 8; ++j) {
          int row = bm + wm * 32 + mi * 16 + (lane >> 4) * 8 + j;
          int col = bn + wn * 64 + ni * 16 + (lane & 15);
          if (!guard || col < N) {
            float v = acc[mi][ni][j];
            size_t o = (size_t)row * N + col;
            if (MODE == 0) {
              Cf[o] = v;
            } else if (MODE == 1) {
              Cf[o] = v + Rf[o];
            } else {
              float g = 0.5f * v * (1.0f + erff(v * 0.70710678118654752f));
              Cb[o] = f2bf(g);
            }
          }
        }
  };
  if (fullN) store_tile(false);
  else       store_tile(true);
}

// ---------------------------------------------------------------------------
// dt = softplus(zxbcdt[..., -1] + dt_bias);  dA = exp(dt * (-exp(A_log)))
// (NHEADS == 1)
// ---------------------------------------------------------------------------
__global__ void dt_kernel(const float* __restrict__ zx,
                          const float* __restrict__ dt_bias,
                          const float* __restrict__ A_log,
                          float* __restrict__ dt, float* __restrict__ dA,
                          int rows, int ldz) {
  int i = blockIdx.x * 256 + threadIdx.x;
  if (i >= rows) return;
  float v  = zx[(size_t)i * ldz + (ldz - 1)] + dt_bias[0];
  float sp = (v > 20.f) ? v : log1pf(__expf(v));
  float A  = -__expf(A_log[0]);
  dt[i] = sp;
  dA[i] = __expf(sp * A);
}

// ---------------------------------------------------------------------------
// Causal depthwise conv (width 4) + bias + SiLU over xBC channels of zxbcdt
// ---------------------------------------------------------------------------
__global__ void conv_silu_kernel(const float* __restrict__ zx,
                                 const float* __restrict__ w,
                                 const float* __restrict__ bias,
                                 float* __restrict__ out,
                                 int Ln, int ldz, int choff) {
  int bl  = blockIdx.x;                         // 0..B*L-1
  int c   = blockIdx.y * 256 + threadIdx.x;     // 0..CONV_DIM-1
  int b   = bl / Ln;
  int l   = bl % Ln;
  float acc = bias[c];
  #pragma unroll
  for (int j = 0; j < 4; ++j) {
    int lt = l + j - 3;
    if (lt >= 0)
      acc += w[c * 4 + j] * zx[((size_t)(b * Ln + lt)) * ldz + choff + c];
  }
  acc = acc / (1.f + __expf(-acc));             // SiLU
  out[(size_t)bl * CONV_DIM + c] = acc;
}

// ---------------------------------------------------------------------------
// SSD scan, NHEADS=1, HEADDIM=1536, D_STATE=128.
// Grid (6, B): each workgroup owns one 256-wide slice of p for one batch.
// Each thread owns h[p, 0:128] entirely in VGPRs (128 regs) and walks the
// 2048 timesteps sequentially; B_t / C_t broadcast through LDS each step.
//   h = h*dA_t + (dt_t*x_t)*B_t ;  y[t,p] = <h, C_t> + D*x_t
// ---------------------------------------------------------------------------
__global__ __launch_bounds__(256, 2)
void ssd_scan_kernel(const float* __restrict__ xBC,
                     const float* __restrict__ dt,
                     const float* __restrict__ dA,
                     const float* __restrict__ Dp,
                     float* __restrict__ y, int Ln) {
  const int tid = threadIdx.x;
  const int p   = blockIdx.x * 256 + tid;       // 0..1535
  const int b   = blockIdx.y;
  __shared__ float sB[D_STATE];
  __shared__ float sC[D_STATE];

  float h[D_STATE];
  #pragma unroll
  for (int n = 0; n < D_STATE; ++n) h[n] = 0.f;
  const float Dh = Dp[0];

  for (int t = 0; t < Ln; ++t) {
    const size_t idx = (size_t)b * Ln + t;
    const float* row = xBC + idx * CONV_DIM;
    if (tid < 128) sB[tid]       = row[D_INNER + tid];
    else           sC[tid - 128] = row[D_INNER + D_STATE + (tid - 128)];
    const float xv  = row[p];
    const float dtv = dt[idx];
    const float dAv = dA[idx];
    __syncthreads();

    const float dtx = dtv * xv;
    float acc = 0.f;
    #pragma unroll
    for (int n = 0; n < D_STATE; ++n) {
      h[n] = h[n] * dAv + dtx * sB[n];
      acc += h[n] * sC[n];
    }
    y[idx * D_INNER + p] = acc + Dh * xv;
    __syncthreads();   // protect sB/sC before next step's overwrite
  }
}

// ---------------------------------------------------------------------------
// Launcher
// ---------------------------------------------------------------------------
extern "C" void kernel_launch(void* const* d_in, const int* in_sizes, int n_in,
                              void* d_out, int out_size, void* d_ws,
                              size_t ws_size, hipStream_t stream) {
  (void)in_sizes; (void)n_in; (void)out_size; (void)ws_size;

  const float* x          = (const float*)d_in[0];
  const float* norm1_w    = (const float*)d_in[1];
  const float* in_proj_W  = (const float*)d_in[2];
  const float* conv_w     = (const float*)d_in[3];
  const float* conv_b     = (const float*)d_in[4];
  const float* dt_bias    = (const float*)d_in[5];
  const float* A_log      = (const float*)d_in[6];
  const float* Dparam     = (const float*)d_in[7];
  const float* ssm_norm_w = (const float*)d_in[8];
  const float* out_proj_W = (const float*)d_in[9];
  const float* norm2_w    = (const float*)d_in[10];
  const float* mlp_W1     = (const float*)d_in[11];
  const float* mlp_W2     = (const float*)d_in[12];
  float* out = (float*)d_out;

  // ---- workspace bump allocator (≈ 361 MB total) ----
  char*  base = (char*)d_ws;
  size_t off  = 0;
  auto alloc = [&](size_t bytes) -> void* {
    void* p = base + off;
    off = (off + bytes + 255) & ~(size_t)255;
    return p;
  };
  unsigned short* xn_bf  = (unsigned short*)alloc((size_t)ROWS * D_M * 2);
  unsigned short* Wip_bf = (unsigned short*)alloc((size_t)D_M * D_IN_PROJ * 2);
  unsigned short* Wop_bf = (unsigned short*)alloc((size_t)D_INNER * D_M * 2);
  unsigned short* W1_bf  = (unsigned short*)alloc((size_t)D_M * D_FF * 2);
  unsigned short* W2_bf  = (unsigned short*)alloc((size_t)D_FF * D_M * 2);
  float* zxbcdt = (float*)alloc((size_t)ROWS * D_IN_PROJ * 4);
  float* xBC    = (float*)alloc((size_t)ROWS * CONV_DIM * 4);
  float* dt     = (float*)alloc((size_t)ROWS * 4);
  float* dA     = (float*)alloc((size_t)ROWS * 4);
  float* yscan  = (float*)alloc((size_t)ROWS * D_INNER * 4);
  unsigned short* ygn_bf = (unsigned short*)alloc((size_t)ROWS * D_INNER * 2);
  float* res    = (float*)alloc((size_t)ROWS * D_M * 4);
  unsigned short* h2_bf  = (unsigned short*)alloc((size_t)ROWS * D_M * 2);
  unsigned short* ff_bf  = (unsigned short*)alloc((size_t)ROWS * D_FF * 2);

  const dim3 blk(256);
  auto cvtGrid = [](size_t n) { return dim3((unsigned)((n + 255) / 256)); };

  // ---- weights -> bf16 ----
  f32_to_bf16_kernel<<<cvtGrid((size_t)D_M * D_IN_PROJ), blk, 0, stream>>>(
      in_proj_W, Wip_bf, (size_t)D_M * D_IN_PROJ);
  f32_to_bf16_kernel<<<cvtGrid((size_t)D_INNER * D_M), blk, 0, stream>>>(
      out_proj_W, Wop_bf, (size_t)D_INNER * D_M);
  f32_to_bf16_kernel<<<cvtGrid((size_t)D_M * D_FF), blk, 0, stream>>>(
      mlp_W1, W1_bf, (size_t)D_M * D_FF);
  f32_to_bf16_kernel<<<cvtGrid((size_t)D_FF * D_M), blk, 0, stream>>>(
      mlp_W2, W2_bf, (size_t)D_FF * D_M);

  // ---- norm1 -> bf16 ----
  rmsnorm_kernel<<<dim3(ROWS), blk, 0, stream>>>(
      x, D_M, x, D_M, norm1_w, xn_bf, D_M, 0);

  // ---- in_proj: zxbcdt = xn @ W_ip  (8192 x 768 x 3329) ----
  gemm_wmma_bf16<0><<<dim3((D_IN_PROJ + 127) / 128, ROWS / 128), blk, 0,
                      stream>>>(xn_bf, Wip_bf, zxbcdt, nullptr, nullptr,
                                ROWS, D_IN_PROJ, D_M);

  // ---- dt / dA ----
  dt_kernel<<<dim3(ROWS / 256), blk, 0, stream>>>(
      zxbcdt, dt_bias, A_log, dt, dA, ROWS, D_IN_PROJ);

  // ---- causal depthwise conv + SiLU on xBC channels ----
  conv_silu_kernel<<<dim3(ROWS, CONV_DIM / 256), blk, 0, stream>>>(
      zxbcdt, conv_w, conv_b, xBC, L_N, D_IN_PROJ, D_INNER);

  // ---- SSD scan (register-resident state) ----
  ssd_scan_kernel<<<dim3(D_INNER / 256, B_N), blk, 0, stream>>>(
      xBC, dt, dA, Dparam, yscan, L_N);

  // ---- gated RMSNorm: rmsnorm(y * silu(z)) -> bf16 ----
  rmsnorm_kernel<<<dim3(ROWS), blk, 0, stream>>>(
      yscan, D_INNER, zxbcdt /* z = first 1536 channels */, D_IN_PROJ,
      ssm_norm_w, ygn_bf, D_INNER, 1);

  // ---- out_proj + residual: res = x + ygn @ W_op ----
  gemm_wmma_bf16<1><<<dim3(D_M / 128, ROWS / 128), blk, 0, stream>>>(
      ygn_bf, Wop_bf, res, nullptr, x, ROWS, D_M, D_INNER);

  // ---- norm2 -> bf16 ----
  rmsnorm_kernel<<<dim3(ROWS), blk, 0, stream>>>(
      res, D_M, res, D_M, norm2_w, h2_bf, D_M, 0);

  // ---- MLP1 + exact GELU -> bf16 ----
  gemm_wmma_bf16<2><<<dim3(D_FF / 128, ROWS / 128), blk, 0, stream>>>(
      h2_bf, W1_bf, nullptr, ff_bf, nullptr, ROWS, D_FF, D_M);

  // ---- MLP2 -> fp32 output ----
  gemm_wmma_bf16<0><<<dim3(D_M / 128, ROWS / 128), blk, 0, stream>>>(
      ff_bf, W2_bf, out, nullptr, nullptr, ROWS, D_M, D_FF);
}
